// SAGE_27264452395336
// MI455X (gfx1250) — matrix operand order repaired and checked by reading, not compiled, not run on hardware
//
#include <hip/hip_runtime.h>

// ---------------------------------------------------------------------------
// Problem constants (from reference)
// ---------------------------------------------------------------------------
#define N0_ 1126400
#define N1_ 112640
#define N2_ 11264
#define N3_ 1024
#define E0_ 1126400
#define E1_ 112640
#define E2_ 10240
#define IN_C_ 128
#define HID_ 256
#define OUT_C_ 47
#define OUT_PAD_ 64           // layer-2 weights padded to 4 WMMA N-tiles

typedef __attribute__((ext_vector_type(16))) __bf16 v16bf;
typedef __attribute__((ext_vector_type(8)))  float  v8f;

#define LDP 264               // LDS row pitch in bf16 elems (528 B -> conflict-free)

__device__ __forceinline__ float atomAddF32(float* p, float v) {
    return __hip_atomic_fetch_add(p, v, __ATOMIC_RELAXED, __HIP_MEMORY_SCOPE_AGENT);
}

// ---------------------------------------------------------------------------
// Zero fill (grid-stride)
// ---------------------------------------------------------------------------
__global__ void fill_zero_kernel(float* __restrict__ p, size_t n) {
    size_t i = (size_t)blockIdx.x * blockDim.x + threadIdx.x;
    size_t stride = (size_t)gridDim.x * blockDim.x;
    for (; i < n; i += stride) p[i] = 0.0f;
}

// ---------------------------------------------------------------------------
// fp32 weights [rows, cols] -> bf16 [prows, cols] (zero padding rows)
// ---------------------------------------------------------------------------
__global__ void cvt_w_bf16_kernel(const float* __restrict__ W, __bf16* __restrict__ Wb,
                                  int rows, int cols, int prows) {
    int i = blockIdx.x * blockDim.x + threadIdx.x;
    int total = prows * cols;
    if (i >= total) return;
    int r = i / cols;
    Wb[i] = (r < rows) ? (__bf16)W[i] : (__bf16)0.0f;
}

// ---------------------------------------------------------------------------
// Edge scatter: one block per edge, blockDim == C (coalesced gather lanes),
// native global f32 atomics for the segment sum + degree count.
// ---------------------------------------------------------------------------
__global__ void scatter_add_kernel(const float* __restrict__ src,
                                   const int* __restrict__ ei, int E, int C,
                                   float* __restrict__ agg, float* __restrict__ cnt) {
    int e = blockIdx.x;
    int row = ei[e];
    int col = ei[E + e];
    const float* s = src + (size_t)row * C;
    float*       a = agg + (size_t)col * C;
    int c = threadIdx.x;
    atomAddF32(&a[c], s[c]);
    if (c == 0) atomAddF32(&cnt[col], 1.0f);
}

// ---------------------------------------------------------------------------
// Fused SAGE layer GEMM:
//   out = relu?( (agg/max(cnt,1)) @ Wl^T + bias + xdst @ Wr^T )
// Block = 128 threads = 4 waves. Block owns 16 output rows; each wave owns a
// 16-column tile (blockIdx.y selects which group of 4 tiles).
// Wl/Wr are bf16, row-major [Cout_padded, Cin]. v_wmma_f32_16x16x32_bf16.
// ---------------------------------------------------------------------------
__global__ __launch_bounds__(128)
void sage_wmma_gemm_kernel(const float* __restrict__ agg, const float* __restrict__ cnt,
                           const float* __restrict__ xdst,
                           const __bf16* __restrict__ Wl, const __bf16* __restrict__ Wr,
                           const float* __restrict__ bias,
                           float* __restrict__ out,
                           int Cin, int Cout, int relu) {
    __shared__ __bf16 sAgg[16 * LDP];
    __shared__ __bf16 sDst[16 * LDP];
    __shared__ float  sScale[16];

    const int tid  = threadIdx.x;
    const int lane = tid & 31;
    const int wv   = tid >> 5;
    const int mbase   = blockIdx.x * 16;
    const int colbase = (blockIdx.y * 4 + wv) * 16;
    const int n   = lane & 15;     // N index (also M index for the A fragment)
    const int sel = lane >> 4;     // lane-half selector

    if (tid < 16) sScale[tid] = 1.0f / fmaxf(cnt[mbase + tid], 1.0f);
    __syncthreads();

    // Stage both A operands for this 16-row tile into LDS as bf16.
    for (int i = tid; i < 16 * Cin; i += 128) {
        int r = i / Cin;
        int k = i - r * Cin;
        size_t g = (size_t)(mbase + r) * Cin + k;
        sAgg[r * LDP + k] = (__bf16)(agg[g] * sScale[r]);
        sDst[r * LDP + k] = (__bf16)(xdst[g]);
    }
    __syncthreads();

    // Accumulator seeded with bias (bias depends only on N -> uniform across regs)
    v8f c;
    const int col = colbase + n;
    float bv = (col < Cout) ? bias[col] : 0.0f;
#pragma unroll
    for (int r = 0; r < 8; ++r) c[r] = bv;

    for (int kk = 0; kk < Cin; kk += 32) {
        // ---- A fragment (16x32 bf16 layout): row = n, contiguous K half chosen by sel
        const __bf16* Ar = &sAgg[n * LDP + kk + sel * 16];
        v16bf a;
#pragma unroll
        for (int v = 0; v < 4; ++v) {
            a[2 * v]         = Ar[2 * v];
            a[2 * v + 1]     = Ar[2 * v + 1];
            a[8 + 2 * v]     = Ar[8 + 2 * v];
            a[8 + 2 * v + 1] = Ar[8 + 2 * v + 1];
        }
        // ---- B fragment (32x16 bf16): B[k][n] = Wl[col][k] (row-major, contiguous K)
        const __bf16* Br = &Wl[(size_t)col * Cin + kk + sel * 16];
        v16bf b;
#pragma unroll
        for (int v = 0; v < 16; ++v) b[v] = Br[v];
        c = __builtin_amdgcn_wmma_f32_16x16x32_bf16(false, a, false, b, (short)0, c,
                                                    false, false);

        // ---- second GEMM: xdst @ Wr^T into the same accumulator
        const __bf16* Dr = &sDst[n * LDP + kk + sel * 16];
        v16bf a2;
#pragma unroll
        for (int v = 0; v < 4; ++v) {
            a2[2 * v]         = Dr[2 * v];
            a2[2 * v + 1]     = Dr[2 * v + 1];
            a2[8 + 2 * v]     = Dr[8 + 2 * v];
            a2[8 + 2 * v + 1] = Dr[8 + 2 * v + 1];
        }
        const __bf16* Br2 = &Wr[(size_t)col * Cin + kk + sel * 16];
        v16bf b2;
#pragma unroll
        for (int v = 0; v < 16; ++v) b2[v] = Br2[v];
        c = __builtin_amdgcn_wmma_f32_16x16x32_bf16(false, a2, false, b2, (short)0, c,
                                                    false, false);
    }

    // C/D layout: VGPR r holds M = r + sel*8, N = lane&15
    if (col < Cout) {
#pragma unroll
        for (int r = 0; r < 8; ++r) {
            int row = mbase + sel * 8 + r;
            float v = c[r];
            if (relu) v = fmaxf(v, 0.0f);
            out[(size_t)row * Cout + col] = v;
        }
    }
}

// ---------------------------------------------------------------------------
// In-place log-softmax over 47 logits; one wave32 per row.
// ---------------------------------------------------------------------------
__global__ void log_softmax47_kernel(float* __restrict__ out) {
    int row  = blockIdx.x;
    int lane = threadIdx.x;   // 32 threads
    float* p = out + (size_t)row * OUT_C_;
    float v0 = (lane < OUT_C_) ? p[lane] : -3.4e38f;
    float v1 = (lane + 32 < OUT_C_) ? p[lane + 32] : -3.4e38f;
    float m = fmaxf(v0, v1);
#pragma unroll
    for (int o = 16; o > 0; o >>= 1) m = fmaxf(m, __shfl_xor(m, o, 32));
    float s = ((lane < OUT_C_) ? expf(v0 - m) : 0.0f) +
              ((lane + 32 < OUT_C_) ? expf(v1 - m) : 0.0f);
#pragma unroll
    for (int o = 16; o > 0; o >>= 1) s += __shfl_xor(s, o, 32);
    float l = m + logf(s);
    if (lane < OUT_C_)      p[lane]      = v0 - l;
    if (lane + 32 < OUT_C_) p[lane + 32] = v1 - l;
}

// ---------------------------------------------------------------------------
// Host launcher
// ---------------------------------------------------------------------------
extern "C" void kernel_launch(void* const* d_in, const int* in_sizes, int n_in,
                              void* d_out, int out_size, void* d_ws, size_t ws_size,
                              hipStream_t stream) {
    const float* x   = (const float*)d_in[0];
    const int*   ei0 = (const int*)d_in[1];
    const int*   ei1 = (const int*)d_in[2];
    const int*   ei2 = (const int*)d_in[3];
    const float* Wl0 = (const float*)d_in[4];
    const float* b0  = (const float*)d_in[5];
    const float* Wr0 = (const float*)d_in[6];
    const float* Wl1 = (const float*)d_in[7];
    const float* b1  = (const float*)d_in[8];
    const float* Wr1 = (const float*)d_in[9];
    const float* Wl2 = (const float*)d_in[10];
    const float* b2  = (const float*)d_in[11];
    const float* Wr2 = (const float*)d_in[12];
    float* out = (float*)d_out;

    char* ws = (char*)d_ws;
    size_t off = 0;
    auto alloc = [&](size_t bytes) -> char* {
        char* p = ws + off;
        off += (bytes + 255) & ~(size_t)255;
        return p;
    };

    // --- zeroed region (agg sums + degree counts), contiguous for one fill ---
    float* agg0 = (float*)alloc((size_t)N1_ * IN_C_ * 4);
    float* cnt0 = (float*)alloc((size_t)N1_ * 4);
    float* agg1 = (float*)alloc((size_t)N2_ * HID_ * 4);
    float* cnt1 = (float*)alloc((size_t)N2_ * 4);
    float* agg2 = (float*)alloc((size_t)N3_ * HID_ * 4);
    float* cnt2 = (float*)alloc((size_t)N3_ * 4);
    size_t zero_bytes = (size_t)((char*)cnt2 - (char*)agg0) + (((size_t)N3_ * 4 + 255) & ~(size_t)255);
    (void)cnt2;

    // --- activations ---
    float* h1 = (float*)alloc((size_t)N1_ * HID_ * 4);
    float* h2 = (float*)alloc((size_t)N2_ * HID_ * 4);

    // --- bf16 weight copies (layer 2 padded to OUT_PAD_ rows) ---
    __bf16* wl0 = (__bf16*)alloc((size_t)HID_ * IN_C_ * 2);
    __bf16* wr0 = (__bf16*)alloc((size_t)HID_ * IN_C_ * 2);
    __bf16* wl1 = (__bf16*)alloc((size_t)HID_ * HID_ * 2);
    __bf16* wr1 = (__bf16*)alloc((size_t)HID_ * HID_ * 2);
    __bf16* wl2 = (__bf16*)alloc((size_t)OUT_PAD_ * HID_ * 2);
    __bf16* wr2 = (__bf16*)alloc((size_t)OUT_PAD_ * HID_ * 2);

    // 1) zero accumulators (required every call: graph replays accumulate)
    fill_zero_kernel<<<2048, 256, 0, stream>>>((float*)agg0, zero_bytes / 4);

    // 2) convert weights to bf16
    {
        int t0 = HID_ * IN_C_;
        cvt_w_bf16_kernel<<<(t0 + 255) / 256, 256, 0, stream>>>(Wl0, wl0, HID_, IN_C_, HID_);
        cvt_w_bf16_kernel<<<(t0 + 255) / 256, 256, 0, stream>>>(Wr0, wr0, HID_, IN_C_, HID_);
        int t1 = HID_ * HID_;
        cvt_w_bf16_kernel<<<(t1 + 255) / 256, 256, 0, stream>>>(Wl1, wl1, HID_, HID_, HID_);
        cvt_w_bf16_kernel<<<(t1 + 255) / 256, 256, 0, stream>>>(Wr1, wr1, HID_, HID_, HID_);
        int t2 = OUT_PAD_ * HID_;
        cvt_w_bf16_kernel<<<(t2 + 255) / 256, 256, 0, stream>>>(Wl2, wl2, OUT_C_, HID_, OUT_PAD_);
        cvt_w_bf16_kernel<<<(t2 + 255) / 256, 256, 0, stream>>>(Wr2, wr2, OUT_C_, HID_, OUT_PAD_);
    }

    // 3) layer 0: scatter-mean over E0 edges, then fused WMMA GEMM + ReLU
    scatter_add_kernel<<<E0_, IN_C_, 0, stream>>>(x, ei0, E0_, IN_C_, agg0, cnt0);
    sage_wmma_gemm_kernel<<<dim3(N1_ / 16, HID_ / 64), 128, 0, stream>>>(
        agg0, cnt0, x, wl0, wr0, b0, h1, IN_C_, HID_, 1);

    // 4) layer 1
    scatter_add_kernel<<<E1_, HID_, 0, stream>>>(h1, ei1, E1_, HID_, agg1, cnt1);
    sage_wmma_gemm_kernel<<<dim3(N2_ / 16, HID_ / 64), 128, 0, stream>>>(
        agg1, cnt1, h1, wl1, wr1, b1, h2, HID_, HID_, 1);

    // 5) layer 2 (47 outputs, padded weight tiles; store bounds-checked)
    scatter_add_kernel<<<E2_, HID_, 0, stream>>>(h2, ei2, E2_, HID_, agg2, cnt2);
    sage_wmma_gemm_kernel<<<dim3(N3_ / 16, OUT_PAD_ / 64), 128, 0, stream>>>(
        agg2, cnt2, h2, wl2, wr2, b2, out, HID_, OUT_C_, 0);

    // 6) row-wise log-softmax in place on d_out
    log_softmax47_kernel<<<N3_, 32, 0, stream>>>(out);
}
